// TransformerOnlyDecoder_46385646797661
// MI455X (gfx1250) — matrix operand order repaired and checked by reading, not compiled, not run on hardware
//
#include <hip/hip_runtime.h>
#include <hip/hip_bf16.h>

// ---------------------------------------------------------------------------
// Problem constants (match reference)
// ---------------------------------------------------------------------------
#define BB   16
#define SS   1024
#define DD   512
#define HH   8
#define DVV  64
#define OUTN 60
#define MTOT (BB * SS)          // 16384 rows
#define SLAB (SS * DVV)         // 65536 elements per (b,h) slab

typedef unsigned short u16;
typedef __attribute__((ext_vector_type(16))) __bf16 bf16x16;
typedef __attribute__((ext_vector_type(8)))  float  f32x8;

// ---------------------------------------------------------------------------
// CDNA5 async global->LDS copy (ASYNCcnt path), guarded for toolchain support
// ---------------------------------------------------------------------------
#define GLOBAL_AS __attribute__((address_space(1)))
#define LDS_AS    __attribute__((address_space(3)))
typedef int i32x4 __attribute__((vector_size(16)));

#if defined(__has_builtin)
#if __has_builtin(__builtin_amdgcn_global_load_async_to_lds_b128) && \
    __has_builtin(__builtin_amdgcn_s_wait_asynccnt)
#define HAVE_ASYNC 1
#endif
#endif
#ifndef HAVE_ASYNC
#define HAVE_ASYNC 0
#endif

__device__ __forceinline__ void cp16_async(const u16* g, u16* l) {
#if HAVE_ASYNC
    __builtin_amdgcn_global_load_async_to_lds_b128(
        (GLOBAL_AS i32x4*)g, (LDS_AS i32x4*)l, 0, 0);
#else
    *(uint4*)l = *(const uint4*)g;
#endif
}
__device__ __forceinline__ void async_wait_all() {
#if HAVE_ASYNC
    __builtin_amdgcn_s_wait_asynccnt(0);
#endif
}

// ---------------------------------------------------------------------------
// Helpers
// ---------------------------------------------------------------------------
__device__ __forceinline__ u16 f2bf(float f) {
    unsigned int u = __float_as_uint(f);
    u += 0x7FFFu + ((u >> 16) & 1u);      // round-to-nearest-even
    return (u16)(u >> 16);
}

union FragU { uint4 q[2]; bf16x16 v; };

// A-matrix style fragment: lane holds row; K halves at koff..koff+7 and koff+16..koff+23
__device__ __forceinline__ bf16x16 frag_rowsplit(const u16* p, int koff) {
    FragU f;
    f.q[0] = *(const uint4*)(p + koff);
    f.q[1] = *(const uint4*)(p + koff + 16);
    return f.v;
}
// Contiguous-k fragment (unused here): lane holds column; 16 sequential K halves at p
__device__ __forceinline__ bf16x16 frag_contig(const u16* p) {
    FragU f;
    f.q[0] = *(const uint4*)(p);
    f.q[1] = *(const uint4*)(p + 8);
    return f.v;
}

__device__ __forceinline__ f32x8 wmma_bf16(bf16x16 a, bf16x16 b, f32x8 c) {
    return __builtin_amdgcn_wmma_f32_16x16x32_bf16(
        /*neg_a=*/false, a, /*neg_b=*/false, b,
        /*c_mod=*/(short)0, c, /*reuse_a=*/false, /*reuse_b=*/false);
}

// ---------------------------------------------------------------------------
// Weight convert + transpose via LDS tile: WT[n*512+k] = bf16(W[k*512+n])
// grid (16,16) tiles of 32x32; 256 threads
// ---------------------------------------------------------------------------
__global__ __launch_bounds__(256)
void wt_kernel(const float* __restrict__ W, u16* __restrict__ WT) {
    __shared__ float t[32][33];
    const int tx = threadIdx.x & 31, ty = threadIdx.x >> 5;   // 32 x 8
    const int kn0 = blockIdx.y * 32;   // k tile base
    const int nn0 = blockIdx.x * 32;   // n tile base
    #pragma unroll
    for (int i = 0; i < 4; i++)        // read W[k][n], coalesced in n
        t[ty + i * 8][tx] = W[(size_t)(kn0 + ty + i * 8) * 512 + nn0 + tx];
    __syncthreads();
    #pragma unroll
    for (int i = 0; i < 4; i++)        // write WT[n][k], coalesced in k
        WT[(size_t)(nn0 + ty + i * 8) * 512 + kn0 + tx] = f2bf(t[tx][ty + i * 8]);
}

// ---------------------------------------------------------------------------
// Positional encoding: h = x + pe(s,d); dual f32 + bf16 output
// ---------------------------------------------------------------------------
__global__ void posenc_kernel(const float* __restrict__ X,
                              float* __restrict__ Hf, u16* __restrict__ Hb) {
    int idx = blockIdx.x * 256 + threadIdx.x;        // 0 .. 8388607
    int d = idx & 511;
    int s = (idx >> 9) & 1023;
    int i2 = d & ~1;
    float f   = __expf((float)i2 * (-9.210340371976184f / 512.0f)); // ln(1e4)
    float ang = (float)s * f;
    float pe  = (d & 1) ? __cosf(ang) : __sinf(ang);
    float v = X[idx] + pe;
    Hf[idx] = v;
    Hb[idx] = f2bf(v);
}

// ---------------------------------------------------------------------------
// Tiled WMMA GEMM: C[16384,512] = A(bf16) x Wt(bf16)^T, fused epilogue.
// 256 threads = 8 waves; tile 128(M) x 64(N); K step 32; LDS double-buffered
// via async global->LDS copies overlapped with WMMA of the current tile.
// ---------------------------------------------------------------------------
template<bool BIAS, bool RELU, bool RES, bool OUTF, bool OUTB>
__global__ __launch_bounds__(256)
void gemm512(const u16* __restrict__ A, const u16* __restrict__ WT,
             const float* __restrict__ bias, const float* __restrict__ res,
             float* __restrict__ outF, u16* __restrict__ outB) {
    __shared__ u16 sA[2][128][40];   // 80B row stride: 16B aligned, conflict-free
    __shared__ u16 sB[2][64][40];

    const int tid  = threadIdx.x;
    const int wave = tid >> 5, lane = tid & 31;
    const int half = lane >> 4, cl = lane & 15;
    const int rowBlk = blockIdx.y * 128;
    const int colBlk = blockIdx.x * 64;

    // per-thread staging coordinates (A: two 16B chunks, B: one 16B chunk)
    const int ra0 = tid >> 2,        ca0 = (tid & 3) << 3;          // A chunk 0
    const int ra1 = (tid + 256) >> 2, ca1 = ((tid + 256) & 3) << 3; // A chunk 1
    const int rb  = tid >> 2,        cb  = (tid & 3) << 3;          // B chunk

    const f32x8 zero = {0.f,0.f,0.f,0.f,0.f,0.f,0.f,0.f};
    f32x8 acc[4] = {zero, zero, zero, zero};

    const int lrow  = wave * 16 + cl;
    const int koffA = half << 3;     // 0 or 8
    const int koffB = half << 4;     // 0 or 16

    // prologue: stage tile kb=0 into buffer 0
    cp16_async(&A [(size_t)(rowBlk + ra0) * 512 + ca0], &sA[0][ra0][ca0]);
    cp16_async(&A [(size_t)(rowBlk + ra1) * 512 + ca1], &sA[0][ra1][ca1]);
    cp16_async(&WT[(size_t)(colBlk + rb ) * 512 + cb ], &sB[0][rb ][cb ]);

    int p = 0;
    for (int kb = 0; kb < 512; kb += 32, p ^= 1) {
        async_wait_all();
        __syncthreads();
        if (kb + 32 < 512) {   // stage next tile into the other buffer
            const int kn = kb + 32;
            cp16_async(&A [(size_t)(rowBlk + ra0) * 512 + kn + ca0], &sA[p ^ 1][ra0][ca0]);
            cp16_async(&A [(size_t)(rowBlk + ra1) * 512 + kn + ca1], &sA[p ^ 1][ra1][ca1]);
            cp16_async(&WT[(size_t)(colBlk + rb ) * 512 + kn + cb ], &sB[p ^ 1][rb ][cb ]);
        }
        bf16x16 af = frag_rowsplit(&sA[p][lrow][0], koffA);
        #pragma unroll
        for (int n = 0; n < 4; n++) {
            bf16x16 bf = frag_rowsplit(&sB[p][n * 16 + cl][0], koffA);
            acc[n] = wmma_bf16(af, bf, acc[n]);
        }
    }

    // Epilogue: C layout => lane holds rows j + 8*half, col cl (per tile)
    const int rbase = rowBlk + wave * 16 + (half << 3);
    #pragma unroll
    for (int n = 0; n < 4; n++) {
        int col = colBlk + n * 16 + cl;
        float bv = BIAS ? bias[col] : 0.0f;
        #pragma unroll
        for (int j = 0; j < 8; j++) {
            size_t o = (size_t)(rbase + j) * 512 + col;
            float v = acc[n][j] + bv;
            if (RELU) v = v > 0.f ? v : 0.f;
            if (RES)  v += res[o];
            if (OUTF) outF[o] = v;
            if (OUTB) outB[o] = f2bf(v);
        }
    }
}

// ---------------------------------------------------------------------------
// Flash attention per (b,h) slab: Q,K,V,O are contiguous [1024,64] bf16.
// grid = (128 bh, 8 q-tiles); 8 waves, wave w owns Q rows 16w..16w+15.
// ---------------------------------------------------------------------------
__global__ __launch_bounds__(256)
void attn_kernel(const u16* __restrict__ Qg, const u16* __restrict__ Kg,
                 const u16* __restrict__ Vg, u16* __restrict__ Og) {
    __shared__ u16 sVt[64][40];       // V tile transposed: [d][t], padded
    __shared__ u16 sP[8][16][40];     // per-wave P tile: [row][t], padded

    const int bh   = blockIdx.x;
    const int qblk = blockIdx.y * 128;
    const size_t base = (size_t)bh * SLAB;
    const u16* q = Qg + base;
    const u16* k = Kg + base;
    const u16* v = Vg + base;
    u16*       o = Og + base;

    const int tid  = threadIdx.x;
    const int wave = tid >> 5, lane = tid & 31;
    const int half = lane >> 4, cl = lane & 15;

    // Preload both Q fragments (d contraction split 0..31 / 32..63)
    const int qrow = qblk + wave * 16 + cl;
    bf16x16 qf0 = frag_rowsplit(q + (size_t)qrow * 64, 0  + (half << 3));
    bf16x16 qf1 = frag_rowsplit(q + (size_t)qrow * 64, 32 + (half << 3));

    const f32x8 zero = {0.f,0.f,0.f,0.f,0.f,0.f,0.f,0.f};
    f32x8 oacc[4] = {zero, zero, zero, zero};
    float mrow[8], lrow[8];
    #pragma unroll
    for (int j = 0; j < 8; j++) { mrow[j] = -__builtin_inff(); lrow[j] = 0.f; }
    const float scale = 0.125f;       // 1/sqrt(64)

    for (int tb = 0; tb < 1024; tb += 32) {
        __syncthreads();
        {   // stage V tile transposed: 32(t) x 64(d) -> sVt[d][t]
            int tt = tid >> 3;
            int d0 = (tid & 7) << 3;
            union { uint4 u; u16 h[8]; } tv;
            tv.u = *(const uint4*)(v + (size_t)(tb + tt) * 64 + d0);
            #pragma unroll
            for (int e = 0; e < 8; e++) sVt[d0 + e][tt] = tv.h[e];
        }
        __syncthreads();

        // scores = Q * K^T  (two 16-wide t subtiles, two d k-steps each)
        f32x8 st[2];
        #pragma unroll
        for (int u = 0; u < 2; u++) {
            const u16* krow = k + (size_t)(tb + u * 16 + cl) * 64;
            f32x8 s = zero;
            s = wmma_bf16(qf0, frag_rowsplit(krow, 0  + (half << 3)), s);
            s = wmma_bf16(qf1, frag_rowsplit(krow, 32 + (half << 3)), s);
            st[u] = s;
        }

        // online softmax (row reductions across the 16-lane half via shfl)
        float p0v[8], p1v[8];
        #pragma unroll
        for (int j = 0; j < 8; j++) {
            float s0 = st[0][j] * scale, s1 = st[1][j] * scale;
            float mx = fmaxf(s0, s1);
            #pragma unroll
            for (int m = 1; m < 16; m <<= 1) mx = fmaxf(mx, __shfl_xor(mx, m, 32));
            float mnew = fmaxf(mrow[j], mx);
            float sc = __expf(mrow[j] - mnew);
            float p0 = __expf(s0 - mnew);
            float p1 = __expf(s1 - mnew);
            float rs = p0 + p1;
            #pragma unroll
            for (int m = 1; m < 16; m <<= 1) rs += __shfl_xor(rs, m, 32);
            lrow[j] = lrow[j] * sc + rs;
            mrow[j] = mnew;
            #pragma unroll
            for (int n = 0; n < 4; n++) oacc[n][j] *= sc;
            p0v[j] = p0; p1v[j] = p1;
        }

        // transpose P (C layout -> A layout) through per-wave LDS
        #pragma unroll
        for (int j = 0; j < 8; j++) {
            int r = j + (half << 3);
            sP[wave][r][cl]      = f2bf(p0v[j]);
            sP[wave][r][cl + 16] = f2bf(p1v[j]);
        }
        __syncthreads();

        // O += P(16x32) * V(32x64)
        bf16x16 pf = frag_rowsplit(&sP[wave][cl][0], half << 3);
        #pragma unroll
        for (int n = 0; n < 4; n++) {
            bf16x16 vf = frag_rowsplit(&sVt[n * 16 + cl][0], half << 3);
            oacc[n] = wmma_bf16(pf, vf, oacc[n]);
        }
    }

    // normalize and store (slab layout == correct [B,S,D] layout)
    #pragma unroll
    for (int j = 0; j < 8; j++) {
        float inv = 1.0f / lrow[j];
        int row = qblk + wave * 16 + j + (half << 3);
        #pragma unroll
        for (int n = 0; n < 4; n++)
            o[(size_t)row * 64 + n * 16 + cl] = f2bf(oacc[n][j] * inv);
    }
}

// ---------------------------------------------------------------------------
// LayerNorm over D=512: one wave per row; dual f32 + bf16 output
// ---------------------------------------------------------------------------
__global__ __launch_bounds__(256)
void ln_kernel(const float* __restrict__ X,
               float* __restrict__ Y, u16* __restrict__ Yb) {
    int row  = blockIdx.x * 8 + (threadIdx.x >> 5);
    int lane = threadIdx.x & 31;
    const float* x = X + (size_t)row * 512;

    float vals[16];
    float s = 0.f, ss = 0.f;
    #pragma unroll
    for (int i = 0; i < 4; i++) {
        float4 v4 = *(const float4*)&x[lane * 4 + i * 128];
        vals[i*4+0]=v4.x; vals[i*4+1]=v4.y; vals[i*4+2]=v4.z; vals[i*4+3]=v4.w;
        s  += v4.x + v4.y + v4.z + v4.w;
        ss += v4.x*v4.x + v4.y*v4.y + v4.z*v4.z + v4.w*v4.w;
    }
    #pragma unroll
    for (int m = 1; m < 32; m <<= 1) {
        s  += __shfl_xor(s,  m, 32);
        ss += __shfl_xor(ss, m, 32);
    }
    float mean = s * (1.0f / 512.0f);
    float var  = ss * (1.0f / 512.0f) - mean * mean;
    float r = rsqrtf(var + 1e-5f);
    #pragma unroll
    for (int i = 0; i < 4; i++)
        #pragma unroll
        for (int e = 0; e < 4; e++) {
            size_t idx = (size_t)row * 512 + lane * 4 + i * 128 + e;
            float y = (vals[i*4+e] - mean) * r;
            Y[idx]  = y;
            Yb[idx] = f2bf(y);
        }
}

// ---------------------------------------------------------------------------
// Output projection: out[16,60] += h[16, 524288] @ w_out[524288, 60]
// Memory-bound on w_out (read exactly once). f32 atomics into zeroed out.
// ---------------------------------------------------------------------------
__global__ void zero_kernel(float* out, int n) {
    int i = blockIdx.x * 256 + threadIdx.x;
    if (i < n) out[i] = 0.f;
}

__global__ __launch_bounds__(256)
void outproj_kernel(const float* __restrict__ Hf, const float* __restrict__ Wout,
                    float* __restrict__ out) {
    __shared__ float sh[16][1024];     // h chunk, 64 KB
    __shared__ float red[4][16][64];   // ty partials, 16 KB

    const int tx = threadIdx.x & 63;   // output column o
    const int ty = threadIdx.x >> 6;   // 0..3 k-stripes
    const size_t kbase = (size_t)blockIdx.x * 1024;

    for (int i = threadIdx.x; i < 16 * 256; i += 256) {  // 4096 float4
        int b = i >> 8, c = (i & 255) << 2;
        *(float4*)&sh[b][c] = *(const float4*)&Hf[(size_t)b * 524288 + kbase + c];
    }
    __syncthreads();

    float acc[16];
    #pragma unroll
    for (int b = 0; b < 16; b++) acc[b] = 0.f;
    const bool act = (tx < OUTN);

    for (int kk = 0; kk < 256; kk++) {
        int k = ty * 256 + kk;
        if ((kk & 31) == 0)   // stream-ahead hint on the 126MB w_out scan
            __builtin_prefetch(&Wout[(kbase + k + 64) * OUTN + tx], 0, 1);
        float w = act ? Wout[(kbase + k) * OUTN + tx] : 0.f;
        #pragma unroll
        for (int b = 0; b < 16; b++) acc[b] += sh[b][k] * w;
    }
    #pragma unroll
    for (int b = 0; b < 16; b++) red[ty][b][tx] = acc[b];
    __syncthreads();
    if (ty == 0 && act) {
        #pragma unroll
        for (int b = 0; b < 16; b++) {
            float v = red[0][b][tx] + red[1][b][tx] + red[2][b][tx] + red[3][b][tx];
            atomicAdd(&out[b * OUTN + tx], v);
        }
    }
}

// ---------------------------------------------------------------------------
// Host launcher
// ---------------------------------------------------------------------------
extern "C" void kernel_launch(void* const* d_in, const int* in_sizes, int n_in,
                              void* d_out, int out_size, void* d_ws, size_t ws_size,
                              hipStream_t stream) {
    const float* x = (const float*)d_in[0];
    const float* w_in[12];
    for (int i = 0; i < 12; i++) w_in[i] = (const float*)d_in[1 + i];
    const float* bias1[2] = {(const float*)d_in[13], (const float*)d_in[15]}; // b1_1, b1_2
    const float* bias2[2] = {(const float*)d_in[14], (const float*)d_in[16]}; // b2_1, b2_2
    const float* w_out = (const float*)d_in[17];
    float* out = (float*)d_out;

    // Workspace carve-up (256B aligned)
    char* ws = (char*)d_ws;
    size_t off = 0;
    auto carve = [&](size_t bytes) -> void* {
        void* p = ws + off;
        off += (bytes + 255) & ~(size_t)255;
        return p;
    };
    const size_t MD = (size_t)MTOT * DD;
    u16*   wT  = (u16*)  carve(12 * 512 * 512 * sizeof(u16)); // transposed bf16 weights
    float* hA  = (float*)carve(MD * sizeof(float));           // activations f32
    u16*   hB  = (u16*)  carve(MD * sizeof(u16));             // activations bf16
    u16*   qB  = (u16*)  carve(MD * sizeof(u16));
    u16*   kB  = (u16*)  carve(MD * sizeof(u16));
    u16*   vB  = (u16*)  carve(MD * sizeof(u16));
    u16*   cB  = (u16*)  carve(MD * sizeof(u16));             // attention context
    float* gG  = (float*)carve(MD * sizeof(float));           // pre-LN f32 scratch
    (void)ws_size; (void)in_sizes; (void)n_in; (void)out_size;

    // 1) weights -> transposed bf16 (LDS tile transpose, coalesced both ways)
    for (int i = 0; i < 12; i++)
        wt_kernel<<<dim3(16, 16), 256, 0, stream>>>(w_in[i], wT + (size_t)i * 262144);

    // 2) positional encoding
    posenc_kernel<<<32768, 256, 0, stream>>>(x, hA, hB);

    const dim3 gGrid(8, 128), gBlk(256);
    for (int blk = 0; blk < 2; blk++) {
        const u16* wq = wT + (size_t)(blk * 6 + 0) * 262144;
        const u16* wk = wT + (size_t)(blk * 6 + 1) * 262144;
        const u16* wv = wT + (size_t)(blk * 6 + 2) * 262144;
        const u16* wo = wT + (size_t)(blk * 6 + 3) * 262144;
        const u16* w1 = wT + (size_t)(blk * 6 + 4) * 262144;
        const u16* w2 = wT + (size_t)(blk * 6 + 5) * 262144;

        // q/k/v projections (bf16 out only)
        gemm512<false,false,false,false,true><<<gGrid, gBlk, 0, stream>>>(hB, wq, nullptr, nullptr, nullptr, qB);
        gemm512<false,false,false,false,true><<<gGrid, gBlk, 0, stream>>>(hB, wk, nullptr, nullptr, nullptr, kB);
        gemm512<false,false,false,false,true><<<gGrid, gBlk, 0, stream>>>(hB, wv, nullptr, nullptr, nullptr, vB);

        // flash attention over 128 (b,h) slabs
        attn_kernel<<<dim3(128, 8), 256, 0, stream>>>(qB, kB, vB, cB);

        // x = h + ctx @ wo  (f32 out)
        gemm512<false,false,true,true,false><<<gGrid, gBlk, 0, stream>>>(cB, wo, nullptr, hA, gG, nullptr);
        // h1 = LN(x)  -> hA(f32) + hB(bf16)
        ln_kernel<<<2048, 256, 0, stream>>>(gG, hA, hB);
        // t = relu(h1 @ w1 + b1)  (bf16 out, reuse qB)
        gemm512<true,true,false,false,true><<<gGrid, gBlk, 0, stream>>>(hB, w1, bias1[blk], nullptr, nullptr, qB);
        // y = t @ w2 + b2 + h1  (f32 out)
        gemm512<true,false,true,true,false><<<gGrid, gBlk, 0, stream>>>(qB, w2, bias2[blk], hA, gG, nullptr);
        // h = LN(y) -> hA + hB
        ln_kernel<<<2048, 256, 0, stream>>>(gG, hA, hB);
    }

    // 3) final projection
    zero_kernel<<<4, 256, 0, stream>>>(out, BB * OUTN);
    outproj_kernel<<<512, 256, 0, stream>>>(hA, w_out, out);
}
